// DecoderRNN_16312285790575
// MI455X (gfx1250) — compile-verified, matrix operand order checked by reading, not run on hardware
//
#include <hip/hip_runtime.h>

// ---------------- problem constants (match reference) ----------------
constexpr int Tc = 30;      // target length
constexpr int Bc = 256;     // batch
constexpr int Hc = 512;     // hidden
constexpr int Vc = 10000;   // vocab
constexpr int Lc = 30;      // encoder length

// ---------------- CDNA5 feature detection ----------------
#ifndef __has_builtin
#define __has_builtin(x) 0
#endif

#if __has_builtin(__builtin_amdgcn_global_load_async_to_lds_b128)
#define USE_ASYNC_LDS 1
#else
#define USE_ASYNC_LDS 0
#endif

#if __has_builtin(__builtin_amdgcn_s_wait_asynccnt)
#define WAIT_ASYNC(n) __builtin_amdgcn_s_wait_asynccnt(n)
#else
#define WAIT_ASYNC(n) asm volatile("s_wait_asynccnt %0" ::"i"(n) : "memory")
#endif

#define AS1 __attribute__((address_space(1)))
#define AS3 __attribute__((address_space(3)))

// ---------------- WMMA / vector types ----------------
typedef __attribute__((ext_vector_type(16))) __bf16 v16bf;
typedef __attribute__((ext_vector_type(8)))  float  v8f;
typedef __attribute__((ext_vector_type(4)))  int    v4i;

union Frag16 {
    v16bf v;
    unsigned int u[8];
};

__device__ __forceinline__ unsigned short f32_to_bf16_rne(float f) {
    unsigned int u = __float_as_uint(f);
    unsigned int r = 0x7FFFu + ((u >> 16) & 1u);
    return (unsigned short)((u + r) >> 16);
}

// ---------------- f32 [K][N] -> bf16 [N][K] tiled transpose ----------------
__global__ __launch_bounds__(256) void transpose_bf16_kernel(
    const float* __restrict__ src, unsigned short* __restrict__ dst, int K, int N)
{
    __shared__ unsigned short tile[32][33];
    const int k0 = blockIdx.y * 32, n0 = blockIdx.x * 32;
    const int tx = threadIdx.x & 31, ty = threadIdx.x >> 5;   // 32 x 8
    #pragma unroll
    for (int r = ty; r < 32; r += 8) {
        const int k = k0 + r, n = n0 + tx;
        float v = (k < K && n < N) ? src[(size_t)k * N + n] : 0.0f;
        tile[r][tx] = f32_to_bf16_rne(v);
    }
    __syncthreads();
    #pragma unroll
    for (int r = ty; r < 32; r += 8) {
        const int n = n0 + r, k = k0 + tx;
        if (n < N && k < K) dst[(size_t)n * K + k] = tile[tx][r];
    }
}

// ---------------- WMMA GEMM ----------------
// C[M,N] = act(concat(A1,A2)[M,K] @ W[K,N] + bias)
// A1 [M,K1] f32, A2 [M,K-K1] f32 row-major; WT = bf16 W^T, [N][K] row-major.
// act: 0=none, 1=relu, 2=sigmoid, 3=tanh
#define BM 64
#define BN 128
#define BKK 32
#define LDA_PAD (BKK + 2)    // sA pad -> 68B row stride (conflict-free)

__global__ __launch_bounds__(256) void gemm_wmma_bf16_kernel(
    const float* __restrict__ A1, const float* __restrict__ A2, int K1, int K,
    const unsigned short* __restrict__ WT, const float* __restrict__ bias,
    float* __restrict__ C, int M, int N, int act)
{
    __shared__ unsigned short sA[BM][LDA_PAD];      // [m][k] bf16, padded
    __shared__ unsigned short sB[2][BN][BKK];       // [n][k] bf16, double-buffered (async dst)

    const int tid  = threadIdx.x;
    const int lane = tid & 31;
    const int wave = tid >> 5;          // 8 waves
    const int wm   = wave & 3;          // 4 strips of 16 rows
    const int wN   = wave >> 2;         // 2 strips of 64 cols
    const int block_m = blockIdx.y * BM;
    const int block_n = blockIdx.x * BN;
    const int K2 = K - K1;

    // B-tile copy geometry: 512 chunks of 16B (8 bf16), 2 per thread.
    const int c0 = tid, c1 = tid + 256;
    const int bn0 = c0 >> 2, bk0 = (c0 & 3) * 8;
    const int bn1 = c1 >> 2, bk1 = (c1 & 3) * 8;
    // clamp column so every wave issues exactly 2 copies per tile (exact ASYNCcnt)
    const int gn0 = min(block_n + bn0, N - 1);
    const int gn1 = min(block_n + bn1, N - 1);

    auto stage_B = [&](int buf, int k0) {
        const unsigned short* g0 = WT + (size_t)gn0 * K + k0 + bk0;
        const unsigned short* g1 = WT + (size_t)gn1 * K + k0 + bk1;
        unsigned short* l0 = &sB[buf][bn0][bk0];
        unsigned short* l1 = &sB[buf][bn1][bk1];
#if USE_ASYNC_LDS
        __builtin_amdgcn_global_load_async_to_lds_b128(
            (AS1 v4i*)(void*)g0, (AS3 v4i*)(void*)l0, 0, 0);
        __builtin_amdgcn_global_load_async_to_lds_b128(
            (AS1 v4i*)(void*)g1, (AS3 v4i*)(void*)l1, 0, 0);
#else
        *(uint4*)l0 = *(const uint4*)g0;
        *(uint4*)l1 = *(const uint4*)g1;
#endif
    };

    v8f acc[4];
    #pragma unroll
    for (int j = 0; j < 4; ++j)
        #pragma unroll
        for (int e = 0; e < 8; ++e) acc[j][e] = 0.0f;

    // A staging geometry: row = tid/4, 8 consecutive k per thread (two float4 loads)
    const int arow = tid >> 2;
    const int akk  = (tid & 3) * 8;
    const int agm  = block_m + arow;

    stage_B(0, 0);                       // prologue: first B tile in flight
    int buf = 0;

    for (int k0 = 0; k0 < K; k0 += BKK) {
        const bool has_next = (k0 + BKK) < K;
        if (has_next) stage_B(buf ^ 1, k0 + BKK);

        // ---- stage A tile (64x32), f32 -> bf16; whole tile from one source ----
        {
            const float* srcA;
            int ld;
            if (k0 < K1) { srcA = A1 + k0;        ld = K1; }
            else         { srcA = A2 + (k0 - K1); ld = K2; }
            if (agm < M) {
                const float4* p = (const float4*)(srcA + (size_t)agm * ld + akk);
                const float4 v0 = p[0], v1 = p[1];
                sA[arow][akk + 0] = f32_to_bf16_rne(v0.x);
                sA[arow][akk + 1] = f32_to_bf16_rne(v0.y);
                sA[arow][akk + 2] = f32_to_bf16_rne(v0.z);
                sA[arow][akk + 3] = f32_to_bf16_rne(v0.w);
                sA[arow][akk + 4] = f32_to_bf16_rne(v1.x);
                sA[arow][akk + 5] = f32_to_bf16_rne(v1.y);
                sA[arow][akk + 6] = f32_to_bf16_rne(v1.z);
                sA[arow][akk + 7] = f32_to_bf16_rne(v1.w);
            } else {
                #pragma unroll
                for (int q = 0; q < 8; ++q) sA[arow][akk + q] = 0;
            }
        }

#if USE_ASYNC_LDS
        if (has_next) WAIT_ASYNC(2); else WAIT_ASYNC(0);
#endif
        __syncthreads();

        // ---- A fragment: 16x32 bf16, ISA layout (K-pairs per VGPR) ----
        Frag16 fa;
        {
            const int m = wm * 16 + (lane & 15);
            #pragma unroll
            for (int v = 0; v < 8; ++v) {
                const int kp = ((v >> 2) << 3) + ((lane >> 4) << 2) + (v & 3);
                fa.u[v] = *(const unsigned int*)&sA[m][kp * 2];
            }
        }
        // ---- B fragments (4x 32x16) + WMMA ----
        #pragma unroll
        for (int j = 0; j < 4; ++j) {
            Frag16 fb;
            const int n = wN * 64 + j * 16 + (lane & 15);
            #pragma unroll
            for (int v = 0; v < 8; ++v) {
                const int kp = ((v >> 2) << 3) + ((lane >> 4) << 2) + (v & 3);
                fb.u[v] = *(const unsigned int*)&sB[buf][n][kp * 2];
            }
            acc[j] = __builtin_amdgcn_wmma_f32_16x16x32_bf16(
                false, fa.v, false, fb.v, (short)0, acc[j], false, false);
        }
        __syncthreads();
        buf ^= 1;
    }

    // ---- epilogue: bias + activation, fp32 store ----
    #pragma unroll
    for (int j = 0; j < 4; ++j) {
        const int gn = block_n + wN * 64 + j * 16 + (lane & 15);
        if (gn >= N) continue;
        const float bv = bias ? bias[gn] : 0.0f;
        #pragma unroll
        for (int r = 0; r < 8; ++r) {
            const int gm = block_m + wm * 16 + ((lane >> 4) << 3) + r;
            if (gm >= M) continue;
            float v = acc[j][r] + bv;
            if      (act == 1) v = fmaxf(v, 0.0f);
            else if (act == 2) v = 1.0f / (1.0f + __expf(-v));
            else if (act == 3) v = tanhf(v);
            C[(size_t)gm * N + gn] = v;
        }
    }
}

// ---------------- attention + context (one block per batch row) ----------------
__global__ __launch_bounds__(128) void attn_ctx_kernel(
    const int* __restrict__ tgt, int t,
    const float* __restrict__ Emb, const float* __restrict__ hx,
    const float* __restrict__ Wattn, const float* __restrict__ battn,
    const float* __restrict__ enc,              // [L,B,H]
    float* __restrict__ emb_out, float* __restrict__ ctx_out)
{
    const int b   = blockIdx.x;
    const int tid = threadIdx.x;
    __shared__ float part[128][Lc];
    __shared__ float aw[Lc];
    __shared__ float awe[Lc];

    const int tok = (t == 0) ? 0 : tgt[(size_t)(t - 1) * Bc + b];
    const float* erow = Emb + (size_t)tok * Hc;

    float p[Lc];
    #pragma unroll
    for (int l = 0; l < Lc; ++l) p[l] = 0.0f;

    for (int k = tid; k < 2 * Hc; k += 128) {
        float a;
        if (k < Hc) {
            a = erow[k];
            emb_out[(size_t)b * Hc + k] = a;
        } else {
            a = hx[(size_t)b * Hc + (k - Hc)];
        }
        const float* wrow = Wattn + (size_t)k * Lc;
        #pragma unroll
        for (int l = 0; l < Lc; ++l) p[l] += a * wrow[l];
    }
    #pragma unroll
    for (int l = 0; l < Lc; ++l) part[tid][l] = p[l];
    __syncthreads();

    if (tid < Lc) {
        float s = battn[tid];
        for (int i = 0; i < 128; ++i) s += part[i][tid];
        aw[tid] = s;
    }
    __syncthreads();
    if (tid < Lc) {
        float m = -1e30f;
        for (int l = 0; l < Lc; ++l) m = fmaxf(m, aw[l]);
        awe[tid] = __expf(aw[tid] - m);
    }
    __syncthreads();
    if (tid < Lc) {
        float s = 0.0f;
        for (int l = 0; l < Lc; ++l) s += awe[l];
        aw[tid] = awe[tid] / s;
    }
    __syncthreads();

    for (int h = tid; h < Hc; h += 128) {
        float s = 0.0f;
        #pragma unroll 5
        for (int l = 0; l < Lc; ++l)
            s += aw[l] * enc[((size_t)l * Bc + b) * Hc + h];
        ctx_out[(size_t)b * Hc + h] = s;
    }
}

// ---------------- block-wide reduce / scan helpers (wave32) ----------------
__device__ __forceinline__ float block_max(float v, float* red, int lane, int wave, int nwaves) {
    #pragma unroll
    for (int off = 16; off > 0; off >>= 1) v = fmaxf(v, __shfl_down(v, off, 32));
    if (lane == 0) red[wave] = v;
    __syncthreads();
    if (wave == 0) {
        float x = (lane < nwaves) ? red[lane] : -3.4e38f;
        #pragma unroll
        for (int off = 16; off > 0; off >>= 1) x = fmaxf(x, __shfl_down(x, off, 32));
        if (lane == 0) red[0] = x;
    }
    __syncthreads();
    float r = red[0];
    __syncthreads();
    return r;
}

__device__ __forceinline__ float block_sum(float v, float* red, int lane, int wave, int nwaves) {
    #pragma unroll
    for (int off = 16; off > 0; off >>= 1) v += __shfl_down(v, off, 32);
    if (lane == 0) red[wave] = v;
    __syncthreads();
    if (wave == 0) {
        float x = (lane < nwaves) ? red[lane] : 0.0f;
        #pragma unroll
        for (int off = 16; off > 0; off >>= 1) x += __shfl_down(x, off, 32);
        if (lane == 0) red[0] = x;
    }
    __syncthreads();
    float r = red[0];
    __syncthreads();
    return r;
}

__device__ __forceinline__ float block_incl_scan(float v, float* red, int lane, int wave, int nwaves) {
    float x = v;
    #pragma unroll
    for (int off = 1; off < 32; off <<= 1) {
        float y = __shfl_up(x, off, 32);
        if (lane >= off) x += y;
    }
    if (lane == 31) red[wave] = x;
    __syncthreads();
    if (wave == 0) {
        float w = (lane < nwaves) ? red[lane] : 0.0f;
        #pragma unroll
        for (int off = 1; off < 32; off <<= 1) {
            float y = __shfl_up(w, off, 32);
            if (lane >= off) w += y;
        }
        if (lane < nwaves) red[lane] = w;
    }
    __syncthreads();
    float r = x + ((wave > 0) ? red[wave - 1] : 0.0f);
    __syncthreads();
    return r;
}

// ---------------- ONLSTM cell: softmax+cumsum over H, elementwise update ----------------
__global__ __launch_bounds__(Hc) void onlstm_cell_kernel(
    const float* __restrict__ f,  const float* __restrict__ i_,
    const float* __restrict__ o,  const float* __restrict__ g,
    const float* __restrict__ ftr, const float* __restrict__ itr,
    const float* __restrict__ cx,
    float* __restrict__ hxn, float* __restrict__ cxn)
{
    const int b = blockIdx.x, tid = threadIdx.x;
    const int lane = tid & 31, wave = tid >> 5;   // 16 waves
    __shared__ float red[16];
    const size_t idx = (size_t)b * Hc + tid;

    const float fv = f[idx],  iv = i_[idx], ov = o[idx], gv = g[idx];
    const float ftv = ftr[idx], itv = itr[idx], cxv = cx[idx];

    float m1 = block_max(ftv, red, lane, wave, 16);
    float e1 = __expf(ftv - m1);
    float s1 = block_sum(e1, red, lane, wave, 16);
    float c1 = block_incl_scan(e1 / s1, red, lane, wave, 16);   // cummax(ft)

    float m2 = block_max(itv, red, lane, wave, 16);
    float e2 = __expf(itv - m2);
    float s2 = block_sum(e2, red, lane, wave, 16);
    float c2 = block_incl_scan(e2 / s2, red, lane, wave, 16);
    float itld = 1.0f - c2;

    float om = c1 * itld;
    float fh = fv * om + (c1 - om);
    float ih = iv * om + (itld - om);
    float cn = fh * cxv + ih * gv;
    cxn[idx] = cn;
    hxn[idx] = ov * tanhf(cn);
}

// ---------------- log-softmax over V, streamed to output ----------------
__global__ __launch_bounds__(256) void log_softmax_kernel(
    const float* __restrict__ logits, float* __restrict__ out)
{
    const int b = blockIdx.x, tid = threadIdx.x;
    const int lane = tid & 31, wave = tid >> 5;   // 8 waves
    __shared__ float red[8];
    const float* row = logits + (size_t)b * Vc;

    float m = -3.4e38f;
    for (int j = tid; j < Vc; j += 256) m = fmaxf(m, row[j]);
    m = block_max(m, red, lane, wave, 8);

    float s = 0.0f;
    for (int j = tid; j < Vc; j += 256) s += __expf(row[j] - m);
    s = block_sum(s, red, lane, wave, 8);

    const float lse = m + logf(s);
    float* orow = out + (size_t)b * Vc;
    for (int j = tid; j < Vc; j += 256) orow[j] = row[j] - lse;
}

// ---------------- launcher ----------------
extern "C" void kernel_launch(void* const* d_in, const int* in_sizes, int n_in,
                              void* d_out, int out_size, void* d_ws, size_t ws_size,
                              hipStream_t stream) {
    (void)in_sizes; (void)n_in; (void)out_size; (void)ws_size;

    const int*   tgt   = (const int*)  d_in[0];
    // d_in[1] = input_variable (unused by reference)
    const float* enc   = (const float*)d_in[2];
    const float* hx0   = (const float*)d_in[3];
    const float* cx0   = (const float*)d_in[4];
    const float* Emb   = (const float*)d_in[5];
    const float* Wattn = (const float*)d_in[6];
    const float* battn = (const float*)d_in[7];
    const float* Wc    = (const float*)d_in[8];  const float* bc  = (const float*)d_in[9];
    const float* Wf    = (const float*)d_in[10]; const float* bf  = (const float*)d_in[11];
    const float* Wi    = (const float*)d_in[12]; const float* bi  = (const float*)d_in[13];
    const float* Wo    = (const float*)d_in[14]; const float* bo  = (const float*)d_in[15];
    const float* Wg    = (const float*)d_in[16]; const float* bg  = (const float*)d_in[17];
    const float* Wft   = (const float*)d_in[18]; const float* bft = (const float*)d_in[19];
    const float* Wit   = (const float*)d_in[20]; const float* bit_= (const float*)d_in[21];
    const float* Wout  = (const float*)d_in[22]; const float* bout= (const float*)d_in[23];
    float* out = (float*)d_out;

    // ---- workspace carving (256B aligned) ----
    char* ws = (char*)d_ws;
    size_t off = 0;
    auto carve = [&](size_t bytes) -> char* {
        char* p = ws + off;
        off += (bytes + 255) & ~(size_t)255;
        return p;
    };
    const size_t gateW = (size_t)2 * Hc * Hc;          // 1024*512 elements
    unsigned short* Wc_b  = (unsigned short*)carve(gateW * 2);   // all stored transposed [N][K]
    unsigned short* Wf_b  = (unsigned short*)carve(gateW * 2);
    unsigned short* Wi_b  = (unsigned short*)carve(gateW * 2);
    unsigned short* Wo_b  = (unsigned short*)carve(gateW * 2);
    unsigned short* Wg_b  = (unsigned short*)carve(gateW * 2);
    unsigned short* Wft_b = (unsigned short*)carve(gateW * 2);
    unsigned short* Wit_b = (unsigned short*)carve(gateW * 2);
    unsigned short* Wout_b= (unsigned short*)carve((size_t)Hc * Vc * 2);

    const size_t bhBytes = (size_t)Bc * Hc * 4;
    float* emb  = (float*)carve(bhBytes);
    float* ctx  = (float*)carve(bhBytes);
    float* xbuf = (float*)carve(bhBytes);
    float* gf   = (float*)carve(bhBytes);
    float* gi   = (float*)carve(bhBytes);
    float* go   = (float*)carve(bhBytes);
    float* gg   = (float*)carve(bhBytes);
    float* gft  = (float*)carve(bhBytes);
    float* git  = (float*)carve(bhBytes);
    float* hxb[2] = { (float*)carve(bhBytes), (float*)carve(bhBytes) };
    float* cxb[2] = { (float*)carve(bhBytes), (float*)carve(bhBytes) };
    float* logits = (float*)carve((size_t)Bc * Vc * 4);

    // ---- transpose+convert weights: f32 [K][N] -> bf16 [N][K] ----
    auto tconv = [&](const float* s, unsigned short* d, int K, int N) {
        dim3 g((N + 31) / 32, (K + 31) / 32);
        transpose_bf16_kernel<<<g, 256, 0, stream>>>(s, d, K, N);
    };
    tconv(Wc,  Wc_b,  2 * Hc, Hc);
    tconv(Wf,  Wf_b,  2 * Hc, Hc);
    tconv(Wi,  Wi_b,  2 * Hc, Hc);
    tconv(Wo,  Wo_b,  2 * Hc, Hc);
    tconv(Wg,  Wg_b,  2 * Hc, Hc);
    tconv(Wft, Wft_b, 2 * Hc, Hc);
    tconv(Wit, Wit_b, 2 * Hc, Hc);
    tconv(Wout, Wout_b, Hc, Vc);

    (void)hipMemcpyAsync(hxb[0], hx0, bhBytes, hipMemcpyDeviceToDevice, stream);
    (void)hipMemcpyAsync(cxb[0], cx0, bhBytes, hipMemcpyDeviceToDevice, stream);

    const dim3 gGate((Hc + BN - 1) / BN, (Bc + BM - 1) / BM);   // 4 x 4
    const dim3 gOut((Vc + BN - 1) / BN, (Bc + BM - 1) / BM);    // 79 x 4

    for (int t = 0; t < Tc; ++t) {
        const int cur = t & 1, nxt = cur ^ 1;

        attn_ctx_kernel<<<Bc, 128, 0, stream>>>(
            tgt, t, Emb, hxb[cur], Wattn, battn, enc, emb, ctx);

        // x = relu([emb, ctx] @ Wc + bc)
        gemm_wmma_bf16_kernel<<<gGate, 256, 0, stream>>>(
            emb, ctx, Hc, 2 * Hc, Wc_b, bc, xbuf, Bc, Hc, 1);

        // gates on xh = [x, hx]
        gemm_wmma_bf16_kernel<<<gGate, 256, 0, stream>>>(xbuf, hxb[cur], Hc, 2 * Hc, Wf_b,  bf,  gf,  Bc, Hc, 2);
        gemm_wmma_bf16_kernel<<<gGate, 256, 0, stream>>>(xbuf, hxb[cur], Hc, 2 * Hc, Wi_b,  bi,  gi,  Bc, Hc, 2);
        gemm_wmma_bf16_kernel<<<gGate, 256, 0, stream>>>(xbuf, hxb[cur], Hc, 2 * Hc, Wo_b,  bo,  go,  Bc, Hc, 2);
        gemm_wmma_bf16_kernel<<<gGate, 256, 0, stream>>>(xbuf, hxb[cur], Hc, 2 * Hc, Wg_b,  bg,  gg,  Bc, Hc, 3);
        gemm_wmma_bf16_kernel<<<gGate, 256, 0, stream>>>(xbuf, hxb[cur], Hc, 2 * Hc, Wft_b, bft, gft, Bc, Hc, 0);
        gemm_wmma_bf16_kernel<<<gGate, 256, 0, stream>>>(xbuf, hxb[cur], Hc, 2 * Hc, Wit_b, bit_,git, Bc, Hc, 0);

        onlstm_cell_kernel<<<Bc, Hc, 0, stream>>>(
            gf, gi, go, gg, gft, git, cxb[cur], hxb[nxt], cxb[nxt]);

        // logits = hx_n @ Wout + bout  (K1 == K, A2 unused)
        gemm_wmma_bf16_kernel<<<gOut, 256, 0, stream>>>(
            hxb[nxt], hxb[nxt], Hc, Hc, Wout_b, bout, logits, Bc, Vc, 0);

        log_softmax_kernel<<<Bc, 256, 0, stream>>>(
            logits, out + (size_t)t * Bc * Vc);
    }
}
